// LinearAttentionBlock_58652073394422
// MI455X (gfx1250) — matrix operand order repaired and checked by reading, not compiled, not run on hardware
//
#include <hip/hip_runtime.h>
#include <hip/hip_bf16.h>

// ---------------------------------------------------------------------------
// Problem constants (reference: B=4, T=1024, Fin=256, D=128)
// ---------------------------------------------------------------------------
#define Bv    4
#define Tv    1024
#define FIN   256
#define Dv    128
#define BT    (Bv * Tv)          // 4096 rows
#define CHK   128                // chunk length
#define NC    (Tv / CHK)         // 8 chunks per batch
#define NCB   (Bv * NC)          // 32 chunks total

typedef __bf16 bf16;
typedef __attribute__((ext_vector_type(16))) __bf16 v16bf;
typedef __attribute__((ext_vector_type(8)))  float  v8f;

// ---------------------------------------------------------------------------
// Async global->LDS copy (CDNA5 GLOBAL_LOAD_ASYNC_TO_LDS_B128, ASYNCcnt).
// Builtin expects: (v4i addrspace(1)*, v4i addrspace(3)*, imm offset, imm cpol)
// Guarded: falls back to a plain b128 round-trip copy if builtin is absent.
// ---------------------------------------------------------------------------
#if defined(__has_builtin)
#if __has_builtin(__builtin_amdgcn_global_load_async_to_lds_b128) && \
    __has_builtin(__builtin_amdgcn_s_wait_asynccnt)
#define USE_ASYNC_LDS 1
#endif
#endif
#ifndef USE_ASYNC_LDS
#define USE_ASYNC_LDS 0
#endif

typedef int v4i_vs __attribute__((vector_size(16)));
typedef __attribute__((address_space(1))) v4i_vs glob_v4i;
typedef __attribute__((address_space(3))) v4i_vs lds_v4i;

__device__ __forceinline__ void cp_lds16(void* lds, const void* g) {
#if USE_ASYNC_LDS
  __builtin_amdgcn_global_load_async_to_lds_b128(
      (glob_v4i*)g, (lds_v4i*)lds, 0, 0);
#else
  *reinterpret_cast<uint4*>(lds) = *reinterpret_cast<const uint4*>(g);
#endif
}
__device__ __forceinline__ void cp_lds_wait() {
#if USE_ASYNC_LDS
  __builtin_amdgcn_s_wait_asynccnt(0);
#endif
}

union FragBF {
  v16bf v;
  bf16  e[16];
  uint4 q[2];
};

__device__ __forceinline__ v8f wmma_bf16(v16bf a, v16bf b, v8f c) {
  // D(f32 16x16) = A(16x32 bf16) * B(32x16 bf16) + C
  return __builtin_amdgcn_wmma_f32_16x16x32_bf16(
      /*neg_a=*/false, a, /*neg_b=*/false, b,
      /*c_mod=*/(short)0, c, /*reuse_a=*/false, /*reuse_b=*/false);
}

// A fragment, row-major source. A[m][k] = src[(row0+m)*ld + k0+k]
// K order used here (contiguous, not the ISA table order): lanes 0-15 -> K 0..15, lanes 16-31 -> K 16..31
__device__ __forceinline__ v16bf load_a_rm(const bf16* src, int ld, int row0,
                                           int k0, int lane) {
  int m = lane & 15, hi = lane >> 4;
  const bf16* p = src + (row0 + m) * ld + k0;
  FragBF f;
  f.q[0] = *reinterpret_cast<const uint4*>(p + hi * 16);
  f.q[1] = *reinterpret_cast<const uint4*>(p + hi * 16 + 8);
  return f.v;
}

// A fragment, transposed source. A[m][k] = src[(k0+k)*ld + m0+m]
__device__ __forceinline__ v16bf load_a_tr(const bf16* src, int ld, int m0,
                                           int k0, int lane) {
  int m = lane & 15, hi = lane >> 4;
  FragBF f;
#pragma unroll
  for (int j = 0; j < 8; ++j) {
    f.e[j]     = src[(k0 + hi * 16 + j) * ld + m0 + m];
    f.e[8 + j] = src[(k0 + hi * 16 + 8 + j) * ld + m0 + m];
  }
  return f.v;
}

// B fragment with contiguous K: B[k][n] = src[(n0+n)*ld + k0+k]
// (i.e. B = srcT; used for weight matrices W[D,Fin] and K^T)
__device__ __forceinline__ v16bf load_b_ct(const bf16* src, int ld, int n0,
                                           int k0, int lane) {
  int n = lane & 15, kh = lane >> 4;
  const bf16* p = src + (n0 + n) * ld + k0 + kh * 16;
  FragBF f;
  f.q[0] = *reinterpret_cast<const uint4*>(p);
  f.q[1] = *reinterpret_cast<const uint4*>(p + 8);
  return f.v;
}

// B fragment, row-major source: B[k][n] = src[(k0+k)*ld + n0+n]
__device__ __forceinline__ v16bf load_b_rm(const bf16* src, int ld, int k0,
                                           int n0, int lane) {
  int n = lane & 15, kh = lane >> 4;
  const bf16* p = src + (k0 + kh * 16) * ld + n0 + n;
  FragBF f;
#pragma unroll
  for (int j = 0; j < 16; ++j) f.e[j] = p[j * ld];
  return f.v;
}

// ---------------------------------------------------------------------------
// K0: fp32 -> bf16 weight conversion
// ---------------------------------------------------------------------------
__global__ void k0_cvt_bf16(const float* __restrict__ src, bf16* __restrict__ dst,
                            int n) {
  int i = blockIdx.x * 256 + threadIdx.x;
  if (i < n) dst[i] = (bf16)src[i];
}

// ---------------------------------------------------------------------------
// K1: fused LayerNorm + {K,Q,V,skip} GEMMs (WMMA bf16)
// One block = 64 rows of [BT, FIN]; 8 waves.
// ---------------------------------------------------------------------------
#define XLD 264  // padded LDS row stride (multiple of 8 -> 16B aligned groups)

__global__ __launch_bounds__(256) void k1_ln_qkvs(
    const float* __restrict__ x, const float* __restrict__ gamma,
    const float* __restrict__ beta,
    const bf16* __restrict__ Wkbf, const bf16* __restrict__ Wqbf,
    const bf16* __restrict__ Wvbf, const bf16* __restrict__ Wsbf,
    float* __restrict__ Kf, float* __restrict__ Vf, float* __restrict__ skipf,
    bf16* __restrict__ Kbf, bf16* __restrict__ Qbf, bf16* __restrict__ Vbf) {
  __shared__ bf16 xn[64 * XLD];
  const int tid = threadIdx.x, lane = tid & 31, wave = tid >> 5;
  const int row0 = blockIdx.x * 64;

  if (tid == 0) {  // pull weights toward L2 (global_prefetch_b8)
    __builtin_prefetch(Wkbf, 0, 2);
    __builtin_prefetch(Wqbf, 0, 2);
    __builtin_prefetch(Wvbf, 0, 2);
    __builtin_prefetch(Wsbf, 0, 2);
  }

  // ---- LayerNorm: each wave handles 8 rows, 8 floats per lane ----
#pragma unroll 1
  for (int rr = 0; rr < 8; ++rr) {
    const int r = wave * 8 + rr;
    const float* xr = x + (size_t)(row0 + r) * FIN;
    float v0[8];
    float s = 0.f;
#pragma unroll
    for (int j = 0; j < 8; ++j) { v0[j] = xr[lane + 32 * j]; s += v0[j]; }
#pragma unroll
    for (int off = 16; off > 0; off >>= 1) s += __shfl_xor(s, off, 32);
    const float mu = s * (1.f / FIN);
    float vs = 0.f;
#pragma unroll
    for (int j = 0; j < 8; ++j) { float d = v0[j] - mu; vs += d * d; }
#pragma unroll
    for (int off = 16; off > 0; off >>= 1) vs += __shfl_xor(vs, off, 32);
    const float rstd = rsqrtf(vs * (1.f / FIN) + 1e-5f);
#pragma unroll
    for (int j = 0; j < 8; ++j) {
      const int f = lane + 32 * j;
      xn[r * XLD + f] = (bf16)((v0[j] - mu) * rstd * gamma[f] + beta[f]);
    }
  }
  __syncthreads();

  // ---- 4 GEMM passes: out[row, d] = sum_f xn[row,f] * W[d,f] ----
  // Each pass: 4x8 = 32 tiles of 16x16, 4 tiles per wave.
#define K1_GEMM_PASS(WPTR, EPILOGUE)                                           \
  _Pragma("unroll 1") for (int t2 = wave; t2 < 32; t2 += 8) {                  \
    const int mt = t2 >> 3, nt = t2 & 7;                                       \
    v8f acc = {};                                                              \
    _Pragma("unroll") for (int kk = 0; kk < FIN / 32; ++kk) {                  \
      v16bf a = load_a_rm(xn, XLD, mt * 16, kk * 32, lane);                    \
      v16bf b = load_b_ct(WPTR, FIN, nt * 16, kk * 32, lane);                  \
      acc = wmma_bf16(a, b, acc);                                              \
    }                                                                          \
    const int n = nt * 16 + (lane & 15), hi = lane >> 4;                       \
    _Pragma("unroll") for (int r = 0; r < 8; ++r) {                            \
      const int row = row0 + mt * 16 + r + 8 * hi;                             \
      const size_t idx = (size_t)row * Dv + n;                                 \
      const float v = acc[r];                                                  \
      EPILOGUE;                                                                \
    }                                                                          \
  }

  // K = elu+1 (fp32 + bf16)
  K1_GEMM_PASS(Wkbf, {
    const float p = v > 0.f ? v + 1.f : __expf(v);
    Kf[idx] = p;
    Kbf[idx] = (bf16)p;
  })
  // Q = elu+1 (bf16)
  K1_GEMM_PASS(Wqbf, {
    const float p = v > 0.f ? v + 1.f : __expf(v);
    Qbf[idx] = (bf16)p;
  })
  // V (fp32 + bf16)
  K1_GEMM_PASS(Wvbf, {
    Vf[idx] = v;
    Vbf[idx] = (bf16)v;
  })
  // skip = xn @ Ws^T (fp32)
  K1_GEMM_PASS(Wsbf, { skipf[idx] = v; })
#undef K1_GEMM_PASS
}

// ---------------------------------------------------------------------------
// K2: per-chunk sums  A[b,c] = K_chunk^T @ V_chunk  (D x D), Zc = sum_t K_t
// One block per chunk (bc = b*NC + c).
// ---------------------------------------------------------------------------
__global__ __launch_bounds__(256) void k2_chunksum(
    const bf16* __restrict__ Kbf, const bf16* __restrict__ Vbf,
    float* __restrict__ A, float* __restrict__ Zc) {
  __shared__ bf16 Kl[CHK * Dv];
  __shared__ bf16 Vl[CHK * Dv];
  const int bc = blockIdx.x, tid = threadIdx.x, lane = tid & 31, wave = tid >> 5;
  const bf16* Kg = Kbf + (size_t)bc * CHK * Dv;
  const bf16* Vg = Vbf + (size_t)bc * CHK * Dv;
#pragma unroll 1
  for (int u = tid; u < (CHK * Dv) / 8; u += 256) {
    cp_lds16(&Kl[u * 8], &Kg[u * 8]);
    cp_lds16(&Vl[u * 8], &Vg[u * 8]);
  }
  cp_lds_wait();
  __syncthreads();

  float* Ab = A + (size_t)bc * Dv * Dv;
#pragma unroll 1
  for (int tix = wave; tix < 64; tix += 8) {
    const int mt = tix >> 3, nt = tix & 7;
    v8f acc = {};
#pragma unroll
    for (int kk = 0; kk < CHK / 32; ++kk) {
      v16bf a = load_a_tr(Kl, Dv, mt * 16, kk * 32, lane);  // A = K^T
      v16bf b = load_b_rm(Vl, Dv, kk * 32, nt * 16, lane);  // B = V
      acc = wmma_bf16(a, b, acc);
    }
    const int n = nt * 16 + (lane & 15), hi = lane >> 4;
#pragma unroll
    for (int r = 0; r < 8; ++r)
      Ab[(mt * 16 + r + 8 * hi) * Dv + n] = acc[r];
  }
  __syncthreads();
  if (tid < Dv) {
    float s = 0.f;
#pragma unroll 1
    for (int t = 0; t < CHK; ++t) s += (float)Kl[t * Dv + tid];
    Zc[bc * Dv + tid] = s;
  }
}

// ---------------------------------------------------------------------------
// K3: exclusive prefix over chunks, seeded by S0 / Z0.
// ---------------------------------------------------------------------------
__global__ void k3_prefix(const float* __restrict__ A, const float* __restrict__ S0,
                          float* __restrict__ P, const float* __restrict__ Zc,
                          const float* __restrict__ Z0, float* __restrict__ Zp) {
  const int idx = blockIdx.x * 256 + threadIdx.x;
  if (idx < Bv * Dv * Dv) {
    const int b = idx >> 14, ij = idx & 16383;
    float acc = S0[idx];
#pragma unroll
    for (int c = 0; c < NC; ++c) {
      const size_t o = ((size_t)(b * NC + c) << 14) + ij;
      P[o] = acc;
      acc += A[o];
    }
  }
  if (idx < Bv * Dv) {
    const int b = idx >> 7, i = idx & 127;
    float acc = Z0[idx];
#pragma unroll
    for (int c = 0; c < NC; ++c) {
      const int o = (b * NC + c) * Dv + i;
      Zp[o] = acc;
      acc += Zc[o];
    }
  }
}

// ---------------------------------------------------------------------------
// K4: intra-chunk rank-1 scan, streams the 256MB S output (float4 stores).
// One block per (bc, j-tile of 32 columns) -> 128 blocks.
// Thread owns a 4x4 (i,j) patch; acc lives in registers.
// ---------------------------------------------------------------------------
__global__ __launch_bounds__(256) void k4_sscan(
    const float* __restrict__ Kf, const float* __restrict__ Vf,
    const float* __restrict__ P, float* __restrict__ Sout) {
  __shared__ float Kl[CHK * Dv];     // 64 KB
  __shared__ float Vl[CHK * 32];     // 16 KB
  const int bid = blockIdx.x, jt = bid & 3, bc = bid >> 2;
  const int tid = threadIdx.x;
  const int j0 = jt * 32;
  const float* Kg = Kf + (size_t)bc * CHK * Dv;
  const float* Vg = Vf + (size_t)bc * CHK * Dv;
#pragma unroll 1
  for (int u = tid; u < (CHK * Dv) / 4; u += 256)
    cp_lds16(&Kl[u * 4], &Kg[u * 4]);
#pragma unroll 1
  for (int u = tid; u < (CHK * 32) / 4; u += 256) {
    const int t = u >> 3, q = u & 7;
    cp_lds16(&Vl[u * 4], Vg + t * Dv + j0 + q * 4);
  }
  cp_lds_wait();
  __syncthreads();

  const int jq = tid & 7, ig = tid >> 3;     // jq: 8 groups of 4 j ; ig: 32 groups of 4 i
  const int i0 = ig * 4, jj0 = jq * 4;
  const float* Pb = P + ((size_t)bc << 14);
  float acc[4][4];
#pragma unroll
  for (int r = 0; r < 4; ++r)
#pragma unroll
    for (int q = 0; q < 4; ++q)
      acc[r][q] = Pb[(i0 + r) * Dv + j0 + jj0 + q];

  float* Sg = Sout + (size_t)bc * CHK * Dv * Dv + j0 + jj0;
#pragma unroll 1
  for (int t = 0; t < CHK; ++t) {
    const float4 v = *(const float4*)(Vl + t * 32 + jj0);
    float kv[4];
#pragma unroll
    for (int r = 0; r < 4; ++r) kv[r] = Kl[t * Dv + i0 + r];
    float* St = Sg + (size_t)t * Dv * Dv;
#pragma unroll
    for (int r = 0; r < 4; ++r) {
      acc[r][0] += kv[r] * v.x;
      acc[r][1] += kv[r] * v.y;
      acc[r][2] += kv[r] * v.z;
      acc[r][3] += kv[r] * v.w;
      float4 st;
      st.x = acc[r][0]; st.y = acc[r][1]; st.z = acc[r][2]; st.w = acc[r][3];
      *(float4*)(St + (i0 + r) * Dv) = st;
    }
  }
}

// ---------------------------------------------------------------------------
// KZ: intra-chunk scan for Z output (tiny: 2MB).
// ---------------------------------------------------------------------------
__global__ void kz_zscan(const float* __restrict__ Kf, const float* __restrict__ Zp,
                         float* __restrict__ Zout) {
  const int idx = blockIdx.x * 256 + threadIdx.x;  // NCB*Dv = 4096 threads
  const int i = idx & 127, bc = idx >> 7;
  float acc = Zp[idx];
  const float* Kg = Kf + (size_t)bc * CHK * Dv + i;
  float* Zg = Zout + (size_t)bc * CHK * Dv + i;
#pragma unroll 1
  for (int t = 0; t < CHK; ++t) {
    acc += Kg[t * Dv];
    Zg[t * Dv] = acc;
  }
}

// ---------------------------------------------------------------------------
// K5: chunked attention numerator/denominator (all WMMA):
//   G = mask(Q K^T);  num = Q @ P + G @ V;  den = Q.Zp + rowsum(G) + eps
//   att = num / den  -> bf16
// One block per chunk.
// ---------------------------------------------------------------------------
__global__ __launch_bounds__(256) void k5_att(
    const bf16* __restrict__ Qbf, const bf16* __restrict__ Kbf,
    const bf16* __restrict__ Vbf, const float* __restrict__ P,
    const float* __restrict__ Zp, bf16* __restrict__ attbf) {
  __shared__ bf16 Ql[CHK * Dv];
  __shared__ bf16 Kl[CHK * Dv];
  __shared__ bf16 Vl[CHK * Dv];
  __shared__ bf16 Pl[Dv * Dv];
  __shared__ bf16 Gl[CHK * CHK];
  __shared__ float denv[CHK];
  const int bc = blockIdx.x, tid = threadIdx.x, lane = tid & 31, wave = tid >> 5;
  const bf16* Qg = Qbf + (size_t)bc * CHK * Dv;
  const bf16* Kg = Kbf + (size_t)bc * CHK * Dv;
  const bf16* Vg = Vbf + (size_t)bc * CHK * Dv;
  const float* Pg = P + ((size_t)bc << 14);
#pragma unroll 1
  for (int u = tid; u < (CHK * Dv) / 8; u += 256) {
    cp_lds16(&Ql[u * 8], &Qg[u * 8]);
    cp_lds16(&Kl[u * 8], &Kg[u * 8]);
    cp_lds16(&Vl[u * 8], &Vg[u * 8]);
  }
#pragma unroll 1
  for (int u = tid; u < Dv * Dv; u += 256) Pl[u] = (bf16)Pg[u];
  cp_lds_wait();
  __syncthreads();

  // den0 = Q_t . Zp + eps
  if (tid < CHK) {
    const float* zp = Zp + bc * Dv;
    float s = 1e-5f;
#pragma unroll 1
    for (int i = 0; i < Dv; ++i) s += (float)Ql[tid * Dv + i] * zp[i];
    denv[tid] = s;
  }

  // G = Q @ K^T with causal mask (inclusive diagonal)
#pragma unroll 1
  for (int tix = wave; tix < 64; tix += 8) {
    const int mt = tix >> 3, nt = tix & 7;
    v8f acc = {};
#pragma unroll
    for (int kk = 0; kk < Dv / 32; ++kk) {
      v16bf a = load_a_rm(Ql, Dv, mt * 16, kk * 32, lane);
      v16bf b = load_b_ct(Kl, Dv, nt * 16, kk * 32, lane);  // B = K^T
      acc = wmma_bf16(a, b, acc);
    }
    const int tcol = nt * 16 + (lane & 15), hi = lane >> 4;
#pragma unroll
    for (int r = 0; r < 8; ++r) {
      const int trow = mt * 16 + r + 8 * hi;
      const float g = (tcol <= trow) ? acc[r] : 0.f;
      Gl[trow * CHK + tcol] = (bf16)g;
    }
  }
  __syncthreads();

  // den = den0 + rowsum(G)
  if (tid < CHK) {
    float s = denv[tid];
#pragma unroll 1
    for (int t2 = 0; t2 < CHK; ++t2) s += (float)Gl[tid * CHK + t2];
    denv[tid] = s;
  }
  __syncthreads();

  // num = Q @ P + G @ V ; att = num / den
#pragma unroll 1
  for (int tix = wave; tix < 64; tix += 8) {
    const int mt = tix >> 3, nt = tix & 7;
    v8f acc = {};
#pragma unroll
    for (int kk = 0; kk < Dv / 32; ++kk) {
      v16bf a = load_a_rm(Ql, Dv, mt * 16, kk * 32, lane);
      v16bf b = load_b_rm(Pl, Dv, kk * 32, nt * 16, lane);
      acc = wmma_bf16(a, b, acc);
    }
#pragma unroll
    for (int kk = 0; kk < CHK / 32; ++kk) {
      v16bf a = load_a_rm(Gl, CHK, mt * 16, kk * 32, lane);
      v16bf b = load_b_rm(Vl, Dv, kk * 32, nt * 16, lane);
      acc = wmma_bf16(a, b, acc);
    }
    const int n = nt * 16 + (lane & 15), hi = lane >> 4;
#pragma unroll
    for (int r = 0; r < 8; ++r) {
      const int trow = mt * 16 + r + 8 * hi;
      attbf[(size_t)bc * CHK * Dv + trow * Dv + n] = (bf16)(acc[r] / denv[trow]);
    }
  }
}

// ---------------------------------------------------------------------------
// K6: 2-layer ReLU MLP (WMMA) + skip + biases -> final out (fp32)
// One block = 64 rows.
// ---------------------------------------------------------------------------
__global__ __launch_bounds__(256) void k6_mlp(
    const bf16* __restrict__ attbf, const bf16* __restrict__ W1bf,
    const bf16* __restrict__ W2bf, const float* __restrict__ b1,
    const float* __restrict__ b2, const float* __restrict__ bs,
    const float* __restrict__ skipf, float* __restrict__ out) {
  __shared__ bf16 h0[64 * Dv];
  __shared__ bf16 h1[64 * Dv];
  const int tid = threadIdx.x, lane = tid & 31, wave = tid >> 5;
  const int row0 = blockIdx.x * 64;
  const bf16* ag = attbf + (size_t)row0 * Dv;
#pragma unroll 1
  for (int u = tid; u < (64 * Dv) / 8; u += 256)
    cp_lds16(&h0[u * 8], &ag[u * 8]);
  cp_lds_wait();
  __syncthreads();

#pragma unroll 1
  for (int tix = wave; tix < 32; tix += 8) {
    const int mt = tix >> 3, nt = tix & 7;
    v8f acc = {};
#pragma unroll
    for (int kk = 0; kk < Dv / 32; ++kk) {
      v16bf a = load_a_rm(h0, Dv, mt * 16, kk * 32, lane);
      v16bf b = load_b_ct(W1bf, Dv, nt * 16, kk * 32, lane);
      acc = wmma_bf16(a, b, acc);
    }
    const int n = nt * 16 + (lane & 15), hi = lane >> 4;
#pragma unroll
    for (int r = 0; r < 8; ++r) {
      const int rw = mt * 16 + r + 8 * hi;
      float v = acc[r] + b1[n];
      h1[rw * Dv + n] = (bf16)(v > 0.f ? v : 0.f);
    }
  }
  __syncthreads();

#pragma unroll 1
  for (int tix = wave; tix < 32; tix += 8) {
    const int mt = tix >> 3, nt = tix & 7;
    v8f acc = {};
#pragma unroll
    for (int kk = 0; kk < Dv / 32; ++kk) {
      v16bf a = load_a_rm(h1, Dv, mt * 16, kk * 32, lane);
      v16bf b = load_b_ct(W2bf, Dv, nt * 16, kk * 32, lane);
      acc = wmma_bf16(a, b, acc);
    }
    const int n = nt * 16 + (lane & 15), hi = lane >> 4;
#pragma unroll
    for (int r = 0; r < 8; ++r) {
      const int rw = mt * 16 + r + 8 * hi;
      float v = acc[r] + b2[n];
      v = v > 0.f ? v : 0.f;
      const size_t idx = (size_t)(row0 + rw) * Dv + n;
      out[idx] = v + skipf[idx] + bs[n];
    }
  }
}

// ---------------------------------------------------------------------------
// Host launcher
// ---------------------------------------------------------------------------
extern "C" void kernel_launch(void* const* d_in, const int* in_sizes, int n_in,
                              void* d_out, int out_size, void* d_ws, size_t ws_size,
                              hipStream_t stream) {
  const float* x     = (const float*)d_in[0];
  const float* S0    = (const float*)d_in[1];
  const float* Z0    = (const float*)d_in[2];
  const float* gamma = (const float*)d_in[3];
  const float* beta  = (const float*)d_in[4];
  const float* Wk    = (const float*)d_in[5];
  const float* Wq    = (const float*)d_in[6];
  const float* Wv    = (const float*)d_in[7];
  const float* W1    = (const float*)d_in[8];
  const float* b1    = (const float*)d_in[9];
  const float* W2    = (const float*)d_in[10];
  const float* b2    = (const float*)d_in[11];
  const float* Ws    = (const float*)d_in[12];
  const float* bs    = (const float*)d_in[13];

  float* out  = (float*)d_out;                       // [B,T,D]
  float* Sout = out + (size_t)BT * Dv;               // [B,T,D,D]
  float* Zout = Sout + (size_t)BT * Dv * Dv;         // [B,T,D]

  char* w = (char*)d_ws;
  size_t off = 0;
  auto alloc = [&](size_t bytes) {
    void* p = w + off;
    off = (off + bytes + 255) & ~(size_t)255;
    return p;
  };
  bf16* Wkbf  = (bf16*)alloc((size_t)Dv * FIN * 2);
  bf16* Wqbf  = (bf16*)alloc((size_t)Dv * FIN * 2);
  bf16* Wvbf  = (bf16*)alloc((size_t)Dv * FIN * 2);
  bf16* Wsbf  = (bf16*)alloc((size_t)Dv * FIN * 2);
  bf16* W1bf  = (bf16*)alloc((size_t)Dv * Dv * 2);
  bf16* W2bf  = (bf16*)alloc((size_t)Dv * Dv * 2);
  bf16* Kbf   = (bf16*)alloc((size_t)BT * Dv * 2);
  bf16* Qbf   = (bf16*)alloc((size_t)BT * Dv * 2);
  bf16* Vbf   = (bf16*)alloc((size_t)BT * Dv * 2);
  bf16* attbf = (bf16*)alloc((size_t)BT * Dv * 2);
  float* Kf    = (float*)alloc((size_t)BT * Dv * 4);
  float* Vf    = (float*)alloc((size_t)BT * Dv * 4);
  float* skipf = (float*)alloc((size_t)BT * Dv * 4);
  float* A     = (float*)alloc((size_t)NCB * Dv * Dv * 4);
  float* P     = (float*)alloc((size_t)NCB * Dv * Dv * 4);
  float* Zc    = (float*)alloc((size_t)NCB * Dv * 4);
  float* Zp    = (float*)alloc((size_t)NCB * Dv * 4);
  (void)ws_size; (void)n_in; (void)in_sizes; (void)out_size;

  // K0: weight conversion
  k0_cvt_bf16<<<(Dv * FIN + 255) / 256, 256, 0, stream>>>(Wk, Wkbf, Dv * FIN);
  k0_cvt_bf16<<<(Dv * FIN + 255) / 256, 256, 0, stream>>>(Wq, Wqbf, Dv * FIN);
  k0_cvt_bf16<<<(Dv * FIN + 255) / 256, 256, 0, stream>>>(Wv, Wvbf, Dv * FIN);
  k0_cvt_bf16<<<(Dv * FIN + 255) / 256, 256, 0, stream>>>(Ws, Wsbf, Dv * FIN);
  k0_cvt_bf16<<<(Dv * Dv + 255) / 256, 256, 0, stream>>>(W1, W1bf, Dv * Dv);
  k0_cvt_bf16<<<(Dv * Dv + 255) / 256, 256, 0, stream>>>(W2, W2bf, Dv * Dv);

  // K1: LN + QKV/skip GEMMs
  k1_ln_qkvs<<<BT / 64, 256, 0, stream>>>(x, gamma, beta, Wkbf, Wqbf, Wvbf, Wsbf,
                                          Kf, Vf, skipf, Kbf, Qbf, Vbf);
  // K2: chunk sums
  k2_chunksum<<<NCB, 256, 0, stream>>>(Kbf, Vbf, A, Zc);
  // K3: prefix over chunks
  k3_prefix<<<(Bv * Dv * Dv + 255) / 256, 256, 0, stream>>>(A, S0, P, Zc, Z0, Zp);
  // K4: S streaming scan (the bandwidth kernel)
  k4_sscan<<<NCB * 4, 256, 0, stream>>>(Kf, Vf, P, Sout);
  // KZ: Z streaming scan
  kz_zscan<<<(NCB * Dv) / 256, 256, 0, stream>>>(Kf, Zp, Zout);
  // K5: chunked attention -> att
  k5_att<<<NCB, 256, 0, stream>>>(Qbf, Kbf, Vbf, P, Zp, attbf);
  // K6: MLP + skip -> out
  k6_mlp<<<BT / 64, 256, 0, stream>>>(attbf, W1bf, W2bf, b1, b2, bs, skipf, out);
}
